// ARC_9715216023714
// MI455X (gfx1250) — matrix operand order repaired and checked by reading, not compiled, not run on hardware
//
#include <hip/hip_runtime.h>

// ---------------------------------------------------------------------------
// ARC forward (Attentive Recurrent Comparator) for MI455X (gfx1250, wave32).
// Heavy math (glimpse batched GEMMs + LSTM gate GEMMs) runs through
// v_wmma_f32_16x16x32_bf16 with fp32 accumulate. GEMMs are templated over
// all strides (compile-time addressing, vectorized float4 staging paths).
// The two LSTM gate GEMMs (x@W_ih^T and Hx@W_hh^T) are fused into a single
// kernel accumulating in WMMA registers — no gates read-modify-write pass.
// ---------------------------------------------------------------------------

typedef __attribute__((ext_vector_type(16))) __bf16 v16bf;
typedef __attribute__((ext_vector_type(8)))  float  v8f;
typedef __attribute__((ext_vector_type(4)))  float  f32x4;
typedef __attribute__((ext_vector_type(4)))  __bf16 bf16x4;

#define B_SZ   256
#define IMG    256
#define GL     64
#define HID    512
#define NSTEP  16

// ---------------------------------------------------------------------------
__global__ void arc_zero_kernel(float* __restrict__ p, int n) {
    int i = blockIdx.x * blockDim.x + threadIdx.x;
    if (i < n) p[i] = 0.0f;
}

// ---------------------------------------------------------------------------
// gp = tanh(Hx @ W_g^T + b_g)   : (256,512)x(512,3) -> (256,3)
// ---------------------------------------------------------------------------
__global__ void arc_controller_kernel(const float* __restrict__ Hx,
                                      const float* __restrict__ Wg,
                                      const float* __restrict__ bg,
                                      float* __restrict__ gp) {
    int b = blockIdx.x;
    int lane = threadIdx.x;
    const float* h = Hx + (long)b * HID;
#pragma unroll
    for (int j = 0; j < 3; ++j) {
        float s = 0.0f;
        for (int k = lane; k < HID; k += 32)
            s += h[k] * Wg[j * HID + k];
#pragma unroll
        for (int off = 16; off > 0; off >>= 1)
            s += __shfl_down(s, off, 32);
        if (lane == 0) gp[b * 3 + j] = tanhf(s + bg[j]);
    }
}

// ---------------------------------------------------------------------------
// Cauchy filterbank: F[b][i][g], normalized over i. blockIdx = (b, g, which)
// ---------------------------------------------------------------------------
__global__ void arc_filterbank_kernel(const float* __restrict__ gp,
                                      float* __restrict__ Fh,
                                      float* __restrict__ Fw) {
    int b = blockIdx.x, g = blockIdx.y, which = blockIdx.z;
    int i = threadIdx.x;                        // image pixel 0..255
    float cc = gp[b * 3 + ((which == 0) ? 0 : 1)];
    float dd = gp[b * 3 + 2];
    const float I = (float)IMG, G = (float)GL;
    float center = (I - 1.0f) * (cc + 1.0f) * 0.5f;
    float delta  = (I / G) * (1.0f - fabsf(dd));
    float gamma  = expf(1.0f - 2.0f * fabsf(dd));
    float gpix   = center + delta * ((float)g - (G - 1.0f) * 0.5f);
    float t      = ((float)i - gpix) / gamma;
    float f      = 1.0f / (3.14159265358979f * gamma * (1.0f + t * t));

    __shared__ float red[IMG];
    red[i] = f;
    __syncthreads();
#pragma unroll
    for (int s = IMG / 2; s > 0; s >>= 1) {
        if (i < s) red[i] += red[i + s];
        __syncthreads();
    }
    float out = f / (red[0] + 1e-4f);
    float* F = (which == 0) ? Fh : Fw;
    F[((long)b * IMG + i) * GL + g] = out;
}

// ---------------------------------------------------------------------------
// Shared bf16-WMMA K-loop phase, compile-time strides.
//   acc[m][n] += sum_k A(m,k) * B(k,n)   over a 64x64 block tile
//   A(m,k) = Ab[m*AM + k*AK],  B(k,n) = Bb[k*BK + n*BN]
// Block: 128 threads = 4 wave32, each wave a 32x32 tile (2x2 WMMA frags).
// Fragment layouts: 16-bit A 16x32, B 32x16 with contiguous K = 16*h + j; f32 C/D per CDNA5 ISA §7.12.2.
// Staging paths: AK==1/BK==1 -> float4 along K + bf16x4 LDS store;
//                else (AM==1/BN==1) -> float4 along tile-row + transposed b16.
// ---------------------------------------------------------------------------
#define LDSS 40   // padded k-stride in LDS (bf16 elems); 80B rows keep b64 alignment

template<long AM, long AK, long BK, long BN, int KTOT>
__device__ __forceinline__ void
arc_gemm_phase(const float* __restrict__ Ab, const float* __restrict__ Bb,
               long blockM, long blockN, int tid, int wm, int wn,
               int fm, int hh, __bf16* Atile, __bf16* Btile,
               v8f (&acc)[2][2]) {
    constexpr int NK = KTOT / 32;
    for (int kt = 0; kt < NK; ++kt) {
        long k0 = (long)kt * 32;

        // ---- stage A tile (64 m-rows x 32 k): load phase, then store phase ----
        f32x4 va[4];
        if constexpr (AK == 1) {          // K-contiguous
#pragma unroll
            for (int i = 0; i < 4; ++i) {
                int e4 = tid + i * 128;   // 512 float4 per tile
                int r  = e4 >> 3;
                int k  = (e4 & 7) << 2;
                va[i] = *(const f32x4*)&Ab[(blockM + r) * AM + (k0 + k)];
            }
#pragma unroll
            for (int i = 0; i < 4; ++i) {
                int e4 = tid + i * 128;
                int r  = e4 >> 3;
                int k  = (e4 & 7) << 2;
                bf16x4 w;
                w[0] = (__bf16)va[i][0]; w[1] = (__bf16)va[i][1];
                w[2] = (__bf16)va[i][2]; w[3] = (__bf16)va[i][3];
                *(bf16x4*)&Atile[r * LDSS + k] = w;
            }
        } else {                          // AM==1: row-contiguous
#pragma unroll
            for (int i = 0; i < 4; ++i) {
                int e4 = tid + i * 128;
                int k  = e4 >> 4;
                int r  = (e4 & 15) << 2;
                va[i] = *(const f32x4*)&Ab[(blockM + r) + (k0 + k) * AK];
            }
#pragma unroll
            for (int i = 0; i < 4; ++i) {
                int e4 = tid + i * 128;
                int k  = e4 >> 4;
                int r  = (e4 & 15) << 2;
#pragma unroll
                for (int j = 0; j < 4; ++j)
                    Atile[(r + j) * LDSS + k] = (__bf16)va[i][j];
            }
        }

        // ---- stage B tile (64 n-cols x 32 k) ----
        f32x4 vb[4];
        if constexpr (BK == 1) {          // K-contiguous
#pragma unroll
            for (int i = 0; i < 4; ++i) {
                int e4 = tid + i * 128;
                int r  = e4 >> 3;
                int k  = (e4 & 7) << 2;
                vb[i] = *(const f32x4*)&Bb[(blockN + r) * BN + (k0 + k)];
            }
#pragma unroll
            for (int i = 0; i < 4; ++i) {
                int e4 = tid + i * 128;
                int r  = e4 >> 3;
                int k  = (e4 & 7) << 2;
                bf16x4 w;
                w[0] = (__bf16)vb[i][0]; w[1] = (__bf16)vb[i][1];
                w[2] = (__bf16)vb[i][2]; w[3] = (__bf16)vb[i][3];
                *(bf16x4*)&Btile[r * LDSS + k] = w;
            }
        } else {                          // BN==1: row-contiguous
#pragma unroll
            for (int i = 0; i < 4; ++i) {
                int e4 = tid + i * 128;
                int k  = e4 >> 4;
                int r  = (e4 & 15) << 2;
                vb[i] = *(const f32x4*)&Bb[(blockN + r) + (k0 + k) * BK];
            }
#pragma unroll
            for (int i = 0; i < 4; ++i) {
                int e4 = tid + i * 128;
                int k  = e4 >> 4;
                int r  = (e4 & 15) << 2;
#pragma unroll
                for (int j = 0; j < 4; ++j)
                    Btile[(r + j) * LDSS + k] = (__bf16)vb[i][j];
            }
        }
        __syncthreads();

        // ---- fragments (16-bit A 16x32 / B 32x16, contiguous K order, same in A and B) ----
        v16bf afr[2], bfr[2];
#pragma unroll
        for (int s = 0; s < 2; ++s) {
            const __bf16* ap = &Atile[(wm + s * 16 + fm) * LDSS];
#pragma unroll
            for (int j = 0; j < 16; ++j)
                afr[s][j] = ap[hh * 16 + j];           // K = h*16 + j
            const __bf16* bp = &Btile[(wn + s * 16 + fm) * LDSS + hh * 16];
#pragma unroll
            for (int j = 0; j < 16; ++j)
                bfr[s][j] = bp[j];                     // K = h*16 + j
        }

#pragma unroll
        for (int i = 0; i < 2; ++i)
#pragma unroll
            for (int j = 0; j < 2; ++j)
                acc[i][j] = __builtin_amdgcn_wmma_f32_16x16x32_bf16(
                    /*neg_a=*/false, afr[i], /*neg_b=*/false, bfr[j],
                    /*c_mod=*/(short)0, acc[i][j],
                    /*reuse_a=*/false, /*reuse_b=*/false);
        __syncthreads();
    }
}

// ---------------------------------------------------------------------------
// Generic single-phase GEMM kernel (used for the two glimpse einsums).
// ---------------------------------------------------------------------------
template<long AM, long AK, long BK, long BN, long CM, int KTOT>
__global__ void __launch_bounds__(128)
arc_wmma_gemm_kernel(const float* __restrict__ A, long aB,
                     const float* __restrict__ B, long bB,
                     float* __restrict__ C, long cB) {
    __shared__ __bf16 Atile[64 * LDSS];
    __shared__ __bf16 Btile[64 * LDSS];

    int tid  = threadIdx.x;
    int lane = tid & 31;
    int wave = tid >> 5;
    int wm = (wave >> 1) * 32;
    int wn = (wave & 1) * 32;
    int fm = lane & 15;
    int hh = lane >> 4;

    long b      = blockIdx.z;
    long blockM = (long)blockIdx.y * 64;
    long blockN = (long)blockIdx.x * 64;

    v8f acc[2][2] = {};
    arc_gemm_phase<AM, AK, BK, BN, KTOT>(A + b * aB, B + b * bB,
                                         blockM, blockN, tid, wm, wn,
                                         fm, hh, Atile, Btile, acc);

    // epilogue: f32 C/D layout — VGPR r: M=r+8*(lane>=16), N=lane&15
    float* Cb = C + b * cB;
    long rowBase = blockM + wm + hh * 8;
    long colBase = blockN + wn + fm;
#pragma unroll
    for (int i = 0; i < 2; ++i)
#pragma unroll
        for (int j = 0; j < 2; ++j)
#pragma unroll
            for (int r = 0; r < 8; ++r)
                Cb[(rowBase + i * 16 + r) * CM + (colBase + j * 16)] = acc[i][j][r];
}

// ---------------------------------------------------------------------------
// Fused LSTM gates GEMM: gates = x @ W_ih^T + Hx @ W_hh^T, accumulated in
// WMMA registers across two K-loop phases (K=4096 then K=512). No gates RMW.
// ---------------------------------------------------------------------------
__global__ void __launch_bounds__(128)
arc_lstm_gates_kernel(const float* __restrict__ xg, const float* __restrict__ W_ih,
                      const float* __restrict__ Hx, const float* __restrict__ W_hh,
                      float* __restrict__ gates) {
    __shared__ __bf16 Atile[64 * LDSS];
    __shared__ __bf16 Btile[64 * LDSS];

    int tid  = threadIdx.x;
    int lane = tid & 31;
    int wave = tid >> 5;
    int wm = (wave >> 1) * 32;
    int wn = (wave & 1) * 32;
    int fm = lane & 15;
    int hh = lane >> 4;

    long blockM = (long)blockIdx.y * 64;
    long blockN = (long)blockIdx.x * 64;

    v8f acc[2][2] = {};
    // phase 1: x (256x4096) @ W_ih^T   A: AM=4096,AK=1  B: BK=1,BN=4096
    arc_gemm_phase<4096L, 1L, 1L, 4096L, 4096>(xg, W_ih, blockM, blockN,
                                               tid, wm, wn, fm, hh,
                                               Atile, Btile, acc);
    // phase 2: Hx (256x512) @ W_hh^T   A: AM=512,AK=1   B: BK=1,BN=512
    arc_gemm_phase<512L, 1L, 1L, 512L, 512>(Hx, W_hh, blockM, blockN,
                                            tid, wm, wn, fm, hh,
                                            Atile, Btile, acc);

    long rowBase = blockM + wm + hh * 8;
    long colBase = blockN + wn + fm;
#pragma unroll
    for (int i = 0; i < 2; ++i)
#pragma unroll
        for (int j = 0; j < 2; ++j)
#pragma unroll
            for (int r = 0; r < 8; ++r)
                gates[(rowBase + i * 16 + r) * (4L * HID) + (colBase + j * 16)] =
                    acc[i][j][r];
}

// ---------------------------------------------------------------------------
// LSTM cell update (torch gate order i,f,g,o split along columns of 4*HID)
// ---------------------------------------------------------------------------
__global__ void arc_lstm_pointwise_kernel(const float* __restrict__ gates,
                                          const float* __restrict__ b_ih,
                                          const float* __restrict__ b_hh,
                                          float* __restrict__ Cx,
                                          float* __restrict__ Hx,
                                          float* __restrict__ final_out) {
    int idx = blockIdx.x * blockDim.x + threadIdx.x;   // 0 .. 256*512-1
    int b = idx >> 9;
    int h = idx & (HID - 1);
    const float* gr = gates + (long)b * (4 * HID);
    float gi = gr[h]            + b_ih[h]            + b_hh[h];
    float gf = gr[HID + h]      + b_ih[HID + h]      + b_hh[HID + h];
    float gg = gr[2 * HID + h]  + b_ih[2 * HID + h]  + b_hh[2 * HID + h];
    float go = gr[3 * HID + h]  + b_ih[3 * HID + h]  + b_hh[3 * HID + h];
    float si = 1.0f / (1.0f + expf(-gi));
    float sf = 1.0f / (1.0f + expf(-gf));
    float so = 1.0f / (1.0f + expf(-go));
    float c  = sf * Cx[idx] + si * tanhf(gg);
    float hx = so * tanhf(c);
    Cx[idx] = c;
    Hx[idx] = hx;
    if (final_out) final_out[idx] = hx;
}

// ---------------------------------------------------------------------------
// host-side orchestration: 16 sequential steps, all on `stream`
// ---------------------------------------------------------------------------
extern "C" void kernel_launch(void* const* d_in, const int* in_sizes, int n_in,
                              void* d_out, int out_size, void* d_ws, size_t ws_size,
                              hipStream_t stream) {
    (void)in_sizes; (void)n_in; (void)out_size; (void)ws_size;

    const float* images = (const float*)d_in[0];   // (256, 2, 256, 256)
    const float* W_ih   = (const float*)d_in[1];   // (2048, 4096)
    const float* W_hh   = (const float*)d_in[2];   // (2048, 512)
    const float* b_ih   = (const float*)d_in[3];   // (2048,)
    const float* b_hh   = (const float*)d_in[4];   // (2048,)
    const float* W_g    = (const float*)d_in[5];   // (3, 512)
    const float* b_g    = (const float*)d_in[6];   // (3,)
    float* out = (float*)d_out;                    // (256, 512)

    // workspace layout (floats)
    float* ws    = (float*)d_ws;
    float* Hx    = ws;  ws += B_SZ * HID;            // 131072
    float* Cx    = ws;  ws += B_SZ * HID;            // 131072
    float* gp    = ws;  ws += 1024;                  // (256,3) padded
    float* Fh    = ws;  ws += (long)B_SZ * IMG * GL; // 4.19M
    float* Fw    = ws;  ws += (long)B_SZ * IMG * GL; // 4.19M
    float* G1    = ws;  ws += (long)B_SZ * GL * IMG; // 4.19M
    float* xg    = ws;  ws += (long)B_SZ * GL * GL;  // 1.05M
    float* gates = ws;  ws += (long)B_SZ * 4 * HID;  // 0.52M

    // Hx, Cx are adjacent: zero both in one pass
    arc_zero_kernel<<<(2 * B_SZ * HID + 255) / 256, 256, 0, stream>>>(Hx, 2 * B_SZ * HID);

    for (int t = 0; t < NSTEP; ++t) {
        long p = (long)(t & 1) * IMG * IMG;   // parity selects image of the pair

        arc_controller_kernel<<<B_SZ, 32, 0, stream>>>(Hx, W_g, b_g, gp);

        arc_filterbank_kernel<<<dim3(B_SZ, GL, 2), IMG, 0, stream>>>(gp, Fh, Fw);

        // G1[b] = F_h[b]^T (64x256) x img[b] (256x256)
        //   A(b,m,k)=Fh[b*16384 + k*64 + m] : AM=1, AK=64 (row-contig path)
        //   B(b,k,n)=img[p + b*131072 + k*256 + n] : BK=256, BN=1 (row-contig path)
        arc_wmma_gemm_kernel<1L, 64L, 256L, 1L, 256L, 256>
            <<<dim3(IMG / 64, GL / 64, B_SZ), 128, 0, stream>>>(
                Fh, (long)IMG * GL,
                images + p, 2L * IMG * IMG,
                G1, (long)GL * IMG);

        // x[b] = G1[b] (64x256) x F_w[b] (256x64)
        //   A: AM=256, AK=1 (K-contig)   B: BK=64, BN=1 (row-contig)
        arc_wmma_gemm_kernel<256L, 1L, 64L, 1L, 64L, 256>
            <<<dim3(GL / 64, GL / 64, B_SZ), 128, 0, stream>>>(
                G1, (long)GL * IMG,
                Fw, (long)IMG * GL,
                xg, (long)GL * GL);

        // gates = x @ W_ih^T + Hx @ W_hh^T  (fused, register-accumulated)
        arc_lstm_gates_kernel<<<dim3(4 * HID / 64, B_SZ / 64, 1), 128, 0, stream>>>(
            xg, W_ih, Hx, W_hh, gates);

        arc_lstm_pointwise_kernel<<<(B_SZ * HID + 255) / 256, 256, 0, stream>>>(
            gates, b_ih, b_hh, Cx, Hx, (t == NSTEP - 1) ? out : (float*)nullptr);
    }
}
